// QLayer_Block_8504035246457
// MI455X (gfx1250) — matrix operand order repaired and checked by reading, not compiled, not run on hardware
//
#include <hip/hip_runtime.h>

typedef __attribute__((ext_vector_type(16))) __bf16 v16bf;
typedef __attribute__((ext_vector_type(8)))  __bf16 v8bf;
typedef __attribute__((ext_vector_type(8)))  float  v8f;
typedef __attribute__((ext_vector_type(4)))  int    v4i;

#define LDSK 40  // padded LDS row stride in bf16 elems (multiple of 8 -> 16B aligned chunks)

// ---- CDNA5 async global->LDS staging (guarded; falls back to reg staging) ----
#if defined(__gfx1250__) && __has_builtin(__builtin_amdgcn_global_load_async_to_lds_b128)
#define ASYNC_LDS 1
#else
#define ASYNC_LDS 0
#endif

__device__ __forceinline__ void async_cp16(const __bf16* g, __bf16* l) {
#if ASYNC_LDS
  __builtin_amdgcn_global_load_async_to_lds_b128(
      (__attribute__((address_space(1))) v4i*)g,
      (__attribute__((address_space(3))) v4i*)l, 0, 0);
#else
  (void)g; (void)l;
#endif
}

__device__ __forceinline__ void async_wait0() {
#if ASYNC_LDS
# if __has_builtin(__builtin_amdgcn_s_wait_asynccnt)
  __builtin_amdgcn_s_wait_asynccnt(0);
# else
  asm volatile("s_wait_asynccnt 0" ::: "memory");
# endif
#endif
}

static __device__ __forceinline__ v16bf frag16(const __bf16* lo, const __bf16* hi) {
  v8bf a = *(const v8bf*)lo;
  v8bf b = *(const v8bf*)hi;
  v16bf r;
#pragma unroll
  for (int i = 0; i < 8; ++i) { r[i] = a[i]; r[i + 8] = b[i]; }
  return r;
}

// ---------------------------------------------------------------------------
// Generic GEMM: C[M,N] = A[M,K] (bf16 row-major) * W[N,K]^T (bf16 row-major)
// + bias[N]; optional ReLU, fp32 residual add, fp32 and/or bf16 outputs.
// Requires M%128==0, N%128==0, K%32==0.
// Block: 256 threads = 8 waves (4 M x 2 N), wave tile 32x64, block tile 128x128.
// ---------------------------------------------------------------------------
template<bool RELU, bool RESID, bool OUTF, bool OUTB>
__global__ __launch_bounds__(256)
void gemm_bf16(const __bf16* __restrict__ A, const __bf16* __restrict__ W,
               const float* __restrict__ bias, const float* resid,
               float* outF, __bf16* outB, int M, int N, int K)
{
  __shared__ __align__(16) __bf16 sA[2][128][LDSK];
  __shared__ __align__(16) __bf16 sB[2][128][LDSK];

  const int tid  = threadIdx.x;
  const int lane = tid & 31;
  const int wave = tid >> 5;
  const int waveM = wave & 3;      // 0..3
  const int waveN = wave >> 2;     // 0..1
  const int mBlk = blockIdx.y * 128;
  const int nBlk = blockIdx.x * 128;

  // Global staging: 2 threads per tile row, each covers 16 halves of the 32-wide K slab.
  const int lr = tid >> 1;                 // 0..127
  const int lc = (tid & 1) * 16;           // 0 or 16
  const __bf16* gA = A + (size_t)(mBlk + lr) * K + lc;
  const __bf16* gW = W + (size_t)(nBlk + lr) * K + lc;

#if ASYNC_LDS
  auto stage = [&](int buf, int k0) {
    async_cp16(gA + k0,     &sA[buf][lr][lc]);
    async_cp16(gA + k0 + 8, &sA[buf][lr][lc + 8]);
    async_cp16(gW + k0,     &sB[buf][lr][lc]);
    async_cp16(gW + k0 + 8, &sB[buf][lr][lc + 8]);
  };
#else
  v8bf ra0, ra1, rb0, rb1;
  auto gload = [&](int k0) {
    ra0 = *(const v8bf*)(gA + k0);
    ra1 = *(const v8bf*)(gA + k0 + 8);
    rb0 = *(const v8bf*)(gW + k0);
    rb1 = *(const v8bf*)(gW + k0 + 8);
  };
  auto sstore = [&](int buf) {
    *(v8bf*)&sA[buf][lr][lc]     = ra0;
    *(v8bf*)&sA[buf][lr][lc + 8] = ra1;
    *(v8bf*)&sB[buf][lr][lc]     = rb0;
    *(v8bf*)&sB[buf][lr][lc + 8] = rb1;
  };
#endif

  v8f acc[2][4] = {};

#if ASYNC_LDS
  stage(0, 0);
  async_wait0();
#else
  gload(0);
  sstore(0);
#endif
  __syncthreads();

  const int nK = K >> 5;
  const int mrow = waveM * 32 + (lane & 15);
  const int kha  = (lane >> 4) * 8;    // A frag: chunks [kha..+7] and [kha+16..+23]
  const int ncol = waveN * 64 + (lane & 15);
  const int khb  = (lane >> 4) * 8;    // B frag: chunks [khb..+7] and [khb+16..+23]

  for (int ks = 0; ks < nK; ++ks) {
    const int buf = ks & 1;
#if ASYNC_LDS
    if (ks + 1 < nK) stage(buf ^ 1, (ks + 1) << 5);
#else
    if (ks + 1 < nK) gload((ks + 1) << 5);
#endif

    v16bf af0 = frag16(&sA[buf][mrow][kha],      &sA[buf][mrow][kha + 16]);
    v16bf af1 = frag16(&sA[buf][mrow + 16][kha], &sA[buf][mrow + 16][kha + 16]);
#pragma unroll
    for (int j = 0; j < 4; ++j) {
      v16bf bfj = frag16(&sB[buf][ncol + j * 16][khb], &sB[buf][ncol + j * 16][khb + 16]);
      acc[0][j] = __builtin_amdgcn_wmma_f32_16x16x32_bf16(
          false, af0, false, bfj, (short)0, acc[0][j], false, false);
      acc[1][j] = __builtin_amdgcn_wmma_f32_16x16x32_bf16(
          false, af1, false, bfj, (short)0, acc[1][j], false, false);
    }

#if ASYNC_LDS
    if (ks + 1 < nK) async_wait0();
#else
    if (ks + 1 < nK) sstore(buf ^ 1);
#endif
    __syncthreads();
  }

  // Epilogue. C layout: VGPR v -> row (base + v + (lane>=16 ? 8 : 0)), col = lane&15.
#pragma unroll
  for (int i = 0; i < 2; ++i) {
    const int rbase = mBlk + waveM * 32 + i * 16 + ((lane >> 4) << 3);
#pragma unroll
    for (int j = 0; j < 4; ++j) {
      const int c = nBlk + waveN * 64 + j * 16 + (lane & 15);
      const float bv = bias[c];
#pragma unroll
      for (int v = 0; v < 8; ++v) {
        float val = acc[i][j][v] + bv;
        if (RELU) val = fmaxf(val, 0.0f);
        const size_t idx = (size_t)(rbase + v) * N + c;
        if (RESID) val += resid[idx];
        if (OUTF) outF[idx] = val;
        if (OUTB) outB[idx] = (__bf16)val;
      }
    }
  }
}

// ---------------------------------------------------------------------------
// Patch-mixing GEMM: Out[b,m,d] = sum_n Wm[m,n]*Act[b,n,d] + biasRow[m]
// Wm is pre-padded/zeroed to 256x224 bf16. Act is [128,196,768] bf16.
// Block tile: 64 rows (m) x 128 cols (d); 8 waves (2 M x 4 N), wave tile 32x32.
// K (=n) padded to 224, stepped by 32. A tile staged async; B transposed via regs.
// ---------------------------------------------------------------------------
__global__ __launch_bounds__(256)
void attn_gemm(const __bf16* __restrict__ Wm, const __bf16* __restrict__ Act,
               const float* __restrict__ biasRow, __bf16* __restrict__ Out)
{
  __shared__ __align__(16) __bf16 sA[2][64][LDSK];
  __shared__ __align__(16) __bf16 sB[2][128][LDSK];   // [d][k] (transposed)

  const int tid  = threadIdx.x;
  const int lane = tid & 31;
  const int wave = tid >> 5;
  const int waveM = wave & 1;     // 0..1
  const int waveN = wave >> 1;    // 0..3
  const int b    = blockIdx.z;
  const int mBlk = blockIdx.y * 64;
  const int dBlk = blockIdx.x * 128;

  // A staging: 4 threads per row, 8 halves each.
  const int ar = tid >> 2;            // 0..63
  const int ak = (tid & 3) * 8;       // 0,8,16,24
  const __bf16* gA = Wm + (size_t)(mBlk + ar) * 224 + ak;

  // B staging: one k-row per thread mod 32, 16 d-values each; transpose on LDS store.
  const int bkr = tid & 31;           // k within slab
  const int bd  = (tid >> 5) * 16;    // 0..112
  const __bf16* gB = Act + (size_t)b * 196 * 768 + dBlk + bd;

#if ASYNC_LDS
  auto stageA = [&](int buf, int k0) {
    async_cp16(gA + k0, &sA[buf][ar][ak]);
  };
#else
  v8bf ra;
  auto gloadA = [&](int k0) { ra = *(const v8bf*)(gA + k0); };
  auto sstoreA = [&](int buf) { *(v8bf*)&sA[buf][ar][ak] = ra; };
#endif

  v8bf rb0, rb1;
  auto gloadB = [&](int k0) {
    const int kk = k0 + bkr;
    if (kk < 196) {
      const __bf16* p = gB + (size_t)kk * 768;
      rb0 = *(const v8bf*)p;
      rb1 = *(const v8bf*)(p + 8);
    } else {
      v8bf z = {};
      rb0 = z; rb1 = z;
    }
  };
  auto sstoreB = [&](int buf) {
#pragma unroll
    for (int i = 0; i < 8; ++i) {
      sB[buf][bd + i][bkr]     = rb0[i];
      sB[buf][bd + 8 + i][bkr] = rb1[i];
    }
  };

  v8f acc[2][2] = {};

#if ASYNC_LDS
  stageA(0, 0);
#else
  gloadA(0);
  sstoreA(0);
#endif
  gloadB(0);
  sstoreB(0);
  async_wait0();
  __syncthreads();

  const int nK = 224 >> 5;   // 7
  const int mrow = waveM * 32 + (lane & 15);
  const int kha  = (lane >> 4) * 8;
  const int ncol = waveN * 32 + (lane & 15);
  const int khb  = (lane >> 4) * 8;

  for (int ks = 0; ks < nK; ++ks) {
    const int buf = ks & 1;
    if (ks + 1 < nK) {
#if ASYNC_LDS
      stageA(buf ^ 1, (ks + 1) << 5);
#else
      gloadA((ks + 1) << 5);
#endif
      gloadB((ks + 1) << 5);
    }

    v16bf af0 = frag16(&sA[buf][mrow][kha],      &sA[buf][mrow][kha + 16]);
    v16bf af1 = frag16(&sA[buf][mrow + 16][kha], &sA[buf][mrow + 16][kha + 16]);
#pragma unroll
    for (int j = 0; j < 2; ++j) {
      v16bf bfj = frag16(&sB[buf][ncol + j * 16][khb], &sB[buf][ncol + j * 16][khb + 16]);
      acc[0][j] = __builtin_amdgcn_wmma_f32_16x16x32_bf16(
          false, af0, false, bfj, (short)0, acc[0][j], false, false);
      acc[1][j] = __builtin_amdgcn_wmma_f32_16x16x32_bf16(
          false, af1, false, bfj, (short)0, acc[1][j], false, false);
    }

    if (ks + 1 < nK) {
      sstoreB(buf ^ 1);
#if ASYNC_LDS
      async_wait0();
#else
      sstoreA(buf ^ 1);
#endif
    }
    __syncthreads();
  }

#pragma unroll
  for (int i = 0; i < 2; ++i) {
    const int rbase = mBlk + waveM * 32 + i * 16 + ((lane >> 4) << 3);
#pragma unroll
    for (int j = 0; j < 2; ++j) {
      const int c = dBlk + waveN * 32 + j * 16 + (lane & 15);
#pragma unroll
      for (int v = 0; v < 8; ++v) {
        const int r = rbase + v;
        if (r < 196) {
          const float val = acc[i][j][v] + biasRow[r];
          Out[((size_t)b * 196 + r) * 768 + c] = (__bf16)val;
        }
      }
    }
  }
}

// ---------------------------------------------------------------------------
// Elementwise converters
// ---------------------------------------------------------------------------
__global__ void cvt_f32_bf16(const float* __restrict__ s, __bf16* __restrict__ d, int n)
{
  const int i = blockIdx.x * 256 + threadIdx.x;
  if (i < n) d[i] = (__bf16)s[i];
}

__global__ void pad_attn_w(const float* __restrict__ s, __bf16* __restrict__ d)
{
  const int i = blockIdx.x * 256 + threadIdx.x;   // 256*224 elements
  if (i >= 256 * 224) return;
  const int m = i / 224;
  const int k = i % 224;
  d[i] = (m < 196 && k < 196) ? (__bf16)s[m * 196 + k] : (__bf16)0.0f;
}

// ---------------------------------------------------------------------------
// Launch
// ---------------------------------------------------------------------------
extern "C" void kernel_launch(void* const* d_in, const int* in_sizes, int n_in,
                              void* d_out, int out_size, void* d_ws, size_t ws_size,
                              hipStream_t stream)
{
  (void)in_sizes; (void)n_in; (void)out_size; (void)ws_size;

  const float* x   = (const float*)d_in[0];
  const float* n1w = (const float*)d_in[1];  const float* n1b = (const float*)d_in[2];
  const float* aw  = (const float*)d_in[3];  const float* ab  = (const float*)d_in[4];
  const float* g1w = (const float*)d_in[5];  const float* g1b = (const float*)d_in[6];
  const float* n2w = (const float*)d_in[7];  const float* n2b = (const float*)d_in[8];
  const float* f1w = (const float*)d_in[9];  const float* f1b = (const float*)d_in[10];
  const float* f2w = (const float*)d_in[11]; const float* f2b = (const float*)d_in[12];
  const float* g2w = (const float*)d_in[13]; const float* g2b = (const float*)d_in[14];
  float* out = (float*)d_out;

  const int Mrows = 128 * 196;                 // 25088
  const size_t MD = (size_t)Mrows * 768;       // 19,267,584
  const size_t MH = (size_t)Mrows * 3072;

  __bf16* xb  = (__bf16*)d_ws;
  __bf16* bA  = xb + MD;
  __bf16* bB  = bA + MD;
  __bf16* h5  = bB + MD;
  __bf16* wn1 = h5 + MH;
  __bf16* wg1 = wn1 + (size_t)768 * 768;
  __bf16* wn2 = wg1 + (size_t)768 * 768;
  __bf16* wg2 = wn2 + (size_t)768 * 768;
  __bf16* wf1 = wg2 + (size_t)768 * 768;
  __bf16* wf2 = wf1 + (size_t)3072 * 768;
  __bf16* wat = wf2 + (size_t)3072 * 768;      // 256x224 padded

  const dim3 blk(256);
  auto cvt = [&](const float* s, __bf16* d, size_t n) {
    cvt_f32_bf16<<<dim3((unsigned)((n + 255) / 256)), blk, 0, stream>>>(s, d, (int)n);
  };

  // Stage 0: one-time bf16 conversions
  cvt(x,   xb,  MD);
  cvt(n1w, wn1, (size_t)768 * 768);
  cvt(g1w, wg1, (size_t)768 * 768);
  cvt(n2w, wn2, (size_t)768 * 768);
  cvt(g2w, wg2, (size_t)768 * 768);
  cvt(f1w, wf1, (size_t)3072 * 768);
  cvt(f2w, wf2, (size_t)3072 * 768);
  pad_attn_w<<<dim3(224), blk, 0, stream>>>(aw, wat);

  // S1: h1 = x @ norm1_w^T + norm1_b          -> bA (bf16)
  gemm_bf16<false, false, false, true><<<dim3(6, 196), blk, 0, stream>>>(
      xb, wn1, n1b, nullptr, nullptr, bA, Mrows, 768, 768);

  // S2: h2[b,m,d] = attn_w @ h1_b + attn_b[m] -> bB (bf16)
  attn_gemm<<<dim3(6, 4, 128), blk, 0, stream>>>(wat, bA, ab, bB);

  // S3: x1 = x + (h2 @ gamma1_w^T + gamma1_b) -> out (fp32) and bA (bf16)
  gemm_bf16<false, true, true, true><<<dim3(6, 196), blk, 0, stream>>>(
      bB, wg1, g1b, x, out, bA, Mrows, 768, 768);

  // S4: h4 = x1 @ norm2_w^T + norm2_b         -> bB (bf16)
  gemm_bf16<false, false, false, true><<<dim3(6, 196), blk, 0, stream>>>(
      bA, wn2, n2b, nullptr, nullptr, bB, Mrows, 768, 768);

  // S5: h5 = relu(h4 @ fc1_w^T + fc1_b)       -> h5 (bf16)
  gemm_bf16<true, false, false, true><<<dim3(24, 196), blk, 0, stream>>>(
      bB, wf1, f1b, nullptr, nullptr, h5, Mrows, 3072, 768);

  // S6: h6 = h5 @ fc2_w^T + fc2_b             -> bA (bf16)
  gemm_bf16<false, false, false, true><<<dim3(6, 196), blk, 0, stream>>>(
      h5, wf2, f2b, nullptr, nullptr, bA, Mrows, 768, 3072);

  // S7: out = x1 + (h6 @ gamma2_w^T + gamma2_b)  (resid = out holds x1)
  gemm_bf16<false, true, true, false><<<dim3(6, 196), blk, 0, stream>>>(
      bA, wg2, g2b, out, out, nullptr, Mrows, 768, 768);
}
